// STGCN_L_50165218018156
// MI455X (gfx1250) — matrix operand order repaired and checked by reading, not compiled, not run on hardware
//
#include <hip/hip_runtime.h>

#define EPSV 1e-5f

// ---------------- CDNA5 WMMA types/helpers (wave32) ----------------
typedef __attribute__((ext_vector_type(16))) __bf16 v16bf;
typedef __attribute__((ext_vector_type(8)))  float  v8f;

struct B32x { uint4 lo, hi; };

__device__ __forceinline__ unsigned short f2bf(float f) {
  union { float f; unsigned u; } c; c.f = f;
  unsigned u = c.u + 0x7FFFu + ((c.u >> 16) & 1u);   // round-to-nearest-even
  return (unsigned short)(u >> 16);
}

// A fragment 16x32 bf16: lanes 0-15 row M=lane hold K {0..7,16..23};
// lanes 16-31 same rows hold K {8..15,24..31}.  LDS tile row-major stride 40.
__device__ __forceinline__ v16bf fragA(const unsigned short* s) {
  int lane = threadIdx.x & 31;
  const unsigned short* p = s + (lane & 15) * 40 + (lane >> 4) * 8;
  B32x t; t.lo = *(const uint4*)p; t.hi = *(const uint4*)(p + 16);
  return __builtin_bit_cast(v16bf, t);
}
// B fragment 32x16 bf16 from LDS stored transposed [n][k] (stride 40):
// lane n (0-15) holds K {0..7,16..23} of column n; lane n+16 holds K {8..15,24..31}.
__device__ __forceinline__ v16bf fragB(const unsigned short* s) {
  int lane = threadIdx.x & 31;
  const unsigned short* p = s + (lane & 15) * 40 + (lane >> 4) * 8;
  B32x t; t.lo = *(const uint4*)p; t.hi = *(const uint4*)(p + 16);
  return __builtin_bit_cast(v16bf, t);
}

#define EF_BIAS 1
#define EF_BN   2
#define EF_RES  4
#define EF_RELU 8

// ---------------- Fused conv / GEMM via WMMA -----------------------
// Out[m, ncol] = sum_{cin,dt} W[m, cin*ntaps+dt] * X[cin, to*stride+dt-pad, v]
// (ncol = to*V + v).  Epilogue: +bias, bn, +res, relu per flags.
// Block 256 threads = 8 waves; tile 64(M) x 128(N); K step 32.
// Wave (wr,wc): wr=wave&3 -> M subtile, wc=wave>>2 -> 64-wide N half.
// Each wave: 1 A-fragment reused over 4 B-fragments -> 4 WMMAs per K step.
__global__ __launch_bounds__(256) void k_conv_wmma(
    const float* __restrict__ Wm, const float* __restrict__ Xin,
    float* __restrict__ Out, const float* __restrict__ bias,
    const float* __restrict__ bng, const float* __restrict__ bnb,
    const float* __restrict__ bnm, const float* __restrict__ bnv,
    const float* __restrict__ Res,
    int M, int Cin, int ntaps, int pad, int stride,
    int Tin, int V, int Ncols, int flags)
{
  __shared__ __attribute__((aligned(16))) unsigned short sA[64 * 40];
  __shared__ __attribute__((aligned(16))) unsigned short sB[128 * 40];
  const int tid = threadIdx.x;
  const int m0 = blockIdx.y * 64, n0 = blockIdx.x * 128;
  const int wave = tid >> 5, wr = wave & 3, wc = wave >> 2;
  const int Ktot = Cin * ntaps;

  v8f acc[4]; acc[0] = {}; acc[1] = {}; acc[2] = {}; acc[3] = {};

  // A-loader assignment: row = tid>>2 (64 rows), 8 contiguous k at (tid&3)*8
  const int ar = tid >> 2, ac = (tid & 3) * 8;
  // B-loader assignment: column = tid&127, k-half = tid>>7 (16 k's each)
  const int nl = tid & 127, kh = (tid >> 7) * 16;
  const int ncolL = n0 + nl;
  const int toL = ncolL / V, vL = ncolL - toL * V;

  for (int k0 = 0; k0 < Ktot; k0 += 32) {
    // ---- A: 64x32 weight tile -> LDS bf16 ----
    {
      int m = m0 + ar;
      const float* src = Wm + (long)m * Ktot + k0 + ac;
      if (m < M && (k0 + 32) <= Ktot && (Ktot & 3) == 0) {
        float4 f0 = *(const float4*)(src);         // global_load_b128
        float4 f1 = *(const float4*)(src + 4);
        unsigned short* d = sA + ar * 40 + ac;
        d[0] = f2bf(f0.x); d[1] = f2bf(f0.y); d[2] = f2bf(f0.z); d[3] = f2bf(f0.w);
        d[4] = f2bf(f1.x); d[5] = f2bf(f1.y); d[6] = f2bf(f1.z); d[7] = f2bf(f1.w);
      } else {
#pragma unroll
        for (int i = 0; i < 8; ++i) {
          int k = k0 + ac + i;
          float v = (m < M && k < Ktot) ? Wm[(long)m * Ktot + k] : 0.f;
          sA[ar * 40 + ac + i] = f2bf(v);
        }
      }
    }
    // ---- B: gathered activations, transposed in LDS [n][k] ----
#pragma unroll
    for (int i = 0; i < 16; ++i) {
      int kl = kh + i;
      int kk = k0 + kl;
      float val = 0.f;
      if (ncolL < Ncols && kk < Ktot) {
        int ci = kk / ntaps, dt = kk - ci * ntaps;
        int t = toL * stride + dt - pad;
        if (t >= 0 && t < Tin) val = Xin[((long)ci * Tin + t) * V + vL];
      }
      sB[nl * 40 + kl] = f2bf(val);
    }
    if (k0 + 32 < Ktot && (m0 + ar) < M)           // gfx1250 global_prefetch_b8
      __builtin_prefetch(&Wm[(long)(m0 + ar) * Ktot + k0 + 32], 0, 0);
    __syncthreads();
    v16bf a = fragA(sA + wr * 16 * 40);
#pragma unroll
    for (int s = 0; s < 4; ++s) {
      v16bf b = fragB(sB + (wc * 64 + s * 16) * 40);
      acc[s] = __builtin_amdgcn_wmma_f32_16x16x32_bf16(false, a, false, b,
                                                       (short)0, acc[s], false, false);
    }
    __syncthreads();
  }

  const int lane = tid & 31, hf = lane >> 4, nn = lane & 15;
#pragma unroll
  for (int s = 0; s < 4; ++s) {
    int ncol = n0 + wc * 64 + s * 16 + nn;
    if (ncol >= Ncols) continue;
#pragma unroll
    for (int r = 0; r < 8; ++r) {             // D layout: VGPR r -> M = r + 8*half
      int m = m0 + wr * 16 + hf * 8 + r;
      if (m >= M) continue;
      float v = acc[s][r];
      if (flags & EF_BIAS) v += bias[m];
      if (flags & EF_BN) {
        float sc = bng[m] * rsqrtf(bnv[m] + EPSV);
        v = (v - bnm[m]) * sc + bnb[m];
      }
      long oi = (long)m * Ncols + ncol;
      if (flags & EF_RES) v += Res[oi];
      if (flags & EF_RELU) v = fmaxf(v, 0.f);
      Out[oi] = v;
    }
  }
}

// ---------------- Graph conv via WMMA ------------------------------
// y[(c,t), w] = sum_{k3,v} H[k3, (c,t), v] * Ae[k3, v, w];  bn1 + relu.
// N is only 44 -> 64-wide tile, 2 WMMAs/wave, 8 waves over 64 M rows.
__global__ __launch_bounds__(256) void k_graph_wmma(
    const float* __restrict__ H, const float* __restrict__ Ae,
    float* __restrict__ Out,
    const float* __restrict__ bng, const float* __restrict__ bnb,
    const float* __restrict__ bnm, const float* __restrict__ bnv,
    int Mrows, int Tlen, int slabA)
{
  __shared__ __attribute__((aligned(16))) unsigned short sA[64 * 40];
  __shared__ __attribute__((aligned(16))) unsigned short sB[64 * 40];
  const int tid = threadIdx.x;
  const int m0 = blockIdx.y * 64;
  const int wave = tid >> 5, wm = wave >> 1, wn = wave & 1;
  v8f acc0 = {}; v8f acc1 = {};
  const int nl = tid & 63, kq = tid >> 6;

  for (int kk0 = 0; kk0 < 192; kk0 += 32) {   // 3 slabs x K=64 (44 padded)
    int k3 = kk0 >> 6, kloc0 = kk0 & 63;
#pragma unroll
    for (int i = 0; i < 8; ++i) {
      int e = tid * 8 + i;
      int r = e >> 5, c = e & 31;
      int m = m0 + r, kl = kloc0 + c;
      float v = (m < Mrows && kl < 44) ? H[(long)k3 * slabA + (long)m * 44 + kl] : 0.f;
      sA[r * 40 + c] = f2bf(v);
    }
#pragma unroll
    for (int i = 0; i < 8; ++i) {
      int klq = kq * 8 + i;
      int kl = kloc0 + klq;
      float v = (nl < 44 && kl < 44) ? Ae[k3 * 1936 + kl * 44 + nl] : 0.f;
      sB[nl * 40 + klq] = f2bf(v);
    }
    __syncthreads();
    v16bf a  = fragA(sA + wm * 16 * 40);
    v16bf b0 = fragB(sB + (wn * 32) * 40);
    v16bf b1 = fragB(sB + (wn * 32 + 16) * 40);
    acc0 = __builtin_amdgcn_wmma_f32_16x16x32_bf16(false, a, false, b0, (short)0, acc0, false, false);
    acc1 = __builtin_amdgcn_wmma_f32_16x16x32_bf16(false, a, false, b1, (short)0, acc1, false, false);
    __syncthreads();
  }
  const int lane = tid & 31, hf = lane >> 4, nn = lane & 15;
#pragma unroll
  for (int sub = 0; sub < 2; ++sub) {
    v8f acc = sub ? acc1 : acc0;
    int w = wn * 32 + sub * 16 + nn;
    if (w >= 44) continue;
#pragma unroll
    for (int r = 0; r < 8; ++r) {
      int m = m0 + wm * 16 + hf * 8 + r;
      if (m >= Mrows) continue;
      int ch = m / Tlen;
      float v = acc[r];
      float s = bng[ch] * rsqrtf(bnv[ch] + EPSV);
      v = (v - bnm[ch]) * s + bnb[ch];
      Out[(long)m * 44 + w] = fmaxf(v, 0.f);
    }
  }
}

// ---------------- Small VALU kernels -------------------------------
__global__ void k_stem(const float* __restrict__ X, const float* __restrict__ w,
                       const float* __restrict__ b, const float* __restrict__ g,
                       const float* __restrict__ bb, const float* __restrict__ m,
                       const float* __restrict__ vv, float* __restrict__ H, int total)
{
  int idx = blockIdx.x * 256 + threadIdx.x;
  if (idx >= total) return;
  int t = idx & 511;
  int j = (idx >> 9) % 44;
  int o = (idx / (44 * 512)) % 32;
  int bq = idx / (32 * 44 * 512);
  long x0 = (((long)bq * 2 + 0) * 44 + j) * 512 + t;
  long x1 = (((long)bq * 2 + 1) * 44 + j) * 512 + t;
  float v = w[o * 2 + 0] * X[x0] + w[o * 2 + 1] * X[x1] + b[o];
  v = fmaxf(v, 0.f);
  float s = g[o] * rsqrtf(vv[o] + EPSV);
  H[idx] = (v - m[o]) * s + bb[o];
}

// attention for one (b,t): scores th^T ph /sqrt(32), softmax over k, apply to g
__global__ __launch_bounds__(256) void k_attn(
    const float* __restrict__ TH, const float* __restrict__ PH,
    const float* __restrict__ GG, float* __restrict__ Y)
{
  __shared__ float sth[32 * 44], sph[32 * 44], sg[32 * 44], ss[44 * 44];
  int bt = blockIdx.x; int b = bt >> 9, t = bt & 511;
  int tid = threadIdx.x;
  for (int e = tid; e < 32 * 44; e += 256) {
    int c = e / 44, j = e % 44;
    long gi = (((long)b * 32 + c) * 44 + j) * 512 + t;
    sth[e] = TH[gi]; sph[e] = PH[gi]; sg[e] = GG[gi];
  }
  __syncthreads();
  for (int e = tid; e < 44 * 44; e += 256) {
    int j = e / 44, k = e % 44;
    float s = 0.f;
    for (int c = 0; c < 32; ++c) s += sth[c * 44 + j] * sph[c * 44 + k];
    ss[e] = s * 0.17677669529663687f;
  }
  __syncthreads();
  if (tid < 44) {
    float mx = -1e30f;
    for (int k = 0; k < 44; ++k) mx = fmaxf(mx, ss[tid * 44 + k]);
    float sum = 0.f;
    for (int k = 0; k < 44; ++k) { float e2 = expf(ss[tid * 44 + k] - mx); ss[tid * 44 + k] = e2; sum += e2; }
    float inv = 1.f / sum;
    for (int k = 0; k < 44; ++k) ss[tid * 44 + k] *= inv;
  }
  __syncthreads();
  for (int e = tid; e < 32 * 44; e += 256) {
    int c = e / 44, j = e % 44;
    float s = 0.f;
    for (int k = 0; k < 44; ++k) s += ss[j * 44 + k] * sg[c * 44 + k];
    Y[(((long)b * 32 + c) * 44 + j) * 512 + t] = s;
  }
}

__global__ void k_deform(const float* __restrict__ H, const float* __restrict__ OFF,
                         float* __restrict__ XO, int total)
{
  int idx = blockIdx.x * 256 + threadIdx.x;
  if (idx >= total) return;
  int t = idx & 511;
  int j = (idx >> 9) % 44;
  int c = (idx / (44 * 512)) % 32;
  int b = idx / (32 * 44 * 512);
  long ob = (((long)b * 64 + c) * 44 + j) * 512 + t;
  float oj = OFF[ob];
  float ot = OFF[ob + (long)32 * 44 * 512];
  float pj = fminf(fmaxf((float)j + oj, 0.f), 43.f);
  float pt = fminf(fmaxf((float)t + ot, 0.f), 511.f);
  float j0f = floorf(pj), t0f = floorf(pt);
  float wj = pj - j0f, wt = pt - t0f;
  int j0 = (int)j0f, t0 = (int)t0f;
  int j1 = j0 + 1 > 43 ? 43 : j0 + 1;
  int t1 = t0 + 1 > 511 ? 511 : t0 + 1;
  long hb = ((long)b * 32 + c) * 44 * 512;
  float g00 = H[hb + (long)j0 * 512 + t0];
  float g01 = H[hb + (long)j0 * 512 + t1];
  float g10 = H[hb + (long)j1 * 512 + t0];
  float g11 = H[hb + (long)j1 * 512 + t1];
  XO[idx] = g00 * (1 - wj) * (1 - wt) + g01 * (1 - wj) * wt
          + g10 * wj * (1 - wt) + g11 * wj * wt;
}

__global__ void k_bott3(const float* __restrict__ Xo, const float* __restrict__ w,
                        const float* __restrict__ b, float* __restrict__ Z, int total)
{
  int idx = blockIdx.x * 256 + threadIdx.x;
  if (idx >= total) return;
  int t = idx & 511;
  int j = (idx >> 9) % 44;
  int c3 = (idx / (44 * 512)) % 3;
  int n = idx / (3 * 44 * 512);
  float s = b[c3];
  long base = ((long)n * 32) * 44 * 512 + (long)j * 512 + t;
  for (int c = 0; c < 32; ++c) s += w[c3 * 32 + c] * Xo[base + (long)c * 44 * 512];
  Z[idx] = s;
}

__global__ void k_sig(const float* __restrict__ Z, float* __restrict__ Sg, int total)
{
  int idx = blockIdx.x * 256 + threadIdx.x;
  if (idx >= total) return;
  int t = idx & 511;
  int j = (idx >> 9) % 44;
  int n = idx / (44 * 512);
  float zc[3][5];
#pragma unroll
  for (int c = 0; c < 3; ++c) {
    long base = (((long)n * 3 + c) * 44 + j) * 512;
#pragma unroll
    for (int i = 0; i < 5; ++i) {
      int tt = t + i; if (tt > 511) tt = 511;     // edge pad
      zc[c][i] = Z[base + tt];
    }
  }
  float a1[3] = {}; float a2[9] = {}; float a3[27] = {};
#pragma unroll
  for (int k = 0; k < 4; ++k) {
    float dk[3];
#pragma unroll
    for (int c = 0; c < 3; ++c) dk[c] = zc[c][k + 1] - zc[c][k];
    float b2[9];
#pragma unroll
    for (int i = 0; i < 3; ++i)
#pragma unroll
      for (int j2 = 0; j2 < 3; ++j2) b2[i * 3 + j2] = 0.5f * dk[i] * dk[j2];
#pragma unroll
    for (int i = 0; i < 3; ++i)
#pragma unroll
      for (int j2 = 0; j2 < 3; ++j2)
#pragma unroll
        for (int l = 0; l < 3; ++l)
          a3[i * 9 + j2 * 3 + l] += dk[i] * b2[j2 * 3 + l] * (1.f / 3.f)
                                  + a1[i] * b2[j2 * 3 + l] + a2[i * 3 + j2] * dk[l];
#pragma unroll
    for (int i = 0; i < 3; ++i)
#pragma unroll
      for (int j2 = 0; j2 < 3; ++j2) a2[i * 3 + j2] += b2[i * 3 + j2] + a1[i] * dk[j2];
#pragma unroll
    for (int i = 0; i < 3; ++i) a1[i] += dk[i];
  }
  long cs = (long)44 * 512;
  long ob = (long)n * 39 * cs + (long)j * 512 + t;
#pragma unroll
  for (int i = 0; i < 3; ++i)  Sg[ob + i * cs] = a1[i];
#pragma unroll
  for (int i = 0; i < 9; ++i)  Sg[ob + (3 + i) * cs] = a2[i];
#pragma unroll
  for (int i = 0; i < 27; ++i) Sg[ob + (12 + i) * cs] = a3[i];
}

__global__ void k_build_bx(const float* __restrict__ H, const float* __restrict__ NL,
                           const float* __restrict__ SP,
                           const float* __restrict__ g, const float* __restrict__ bb,
                           const float* __restrict__ m, const float* __restrict__ vv,
                           float* __restrict__ BX, int total)
{
  int idx = blockIdx.x * 256 + threadIdx.x;
  if (idx >= total) return;
  int v = idx % 44;
  int t = (idx / 44) % 512;
  int c = (idx / (44 * 512)) % 64;
  int b = idx / (64 * 512 * 44);
  long si = (((long)b * 32 + (c & 31)) * 44 + v) * 512 + t;
  float s = (c < 32) ? (H[si] + NL[si]) : SP[si];
  int ch = v * 64 + c;
  float sc = g[ch] * rsqrtf(vv[ch] + EPSV);
  BX[idx] = (s - m[ch]) * sc + bb[ch];
}

__device__ const int cEa[21] = {0,1,2,3,4,1,6,7,8,1,10,11,12,1,14,15,16,1,18,19,20};
__device__ const int cEb[21] = {1,2,3,4,5,6,7,8,9,10,11,12,13,14,15,16,17,18,19,20,21};

__global__ void k_make_adj(const float* __restrict__ edge, float* __restrict__ Ae)
{
  int idx = blockIdx.x * 256 + threadIdx.x;
  if (idx >= 3 * 44 * 44) return;
  int k = idx / 1936, r = idx % 1936, v = r / 44, w = r % 44;
  float a = 0.f;
  if ((v < 22) == (w < 22)) {
    int vv = v % 22, ww = w % 22;
    if (k == 1) a = (vv == ww) ? 1.f : 0.f;
    else {
#pragma unroll
      for (int e = 0; e < 21; ++e) {
        if (k == 0 && vv == cEa[e] && ww == cEb[e]) a = 1.f;
        if (k == 2 && vv == cEb[e] && ww == cEa[e]) a = 1.f;
      }
    }
  }
  Ae[idx] = a * edge[idx];
}

__global__ void k_mean(const float* __restrict__ x, float* __restrict__ feat,
                       int C, int T, int V)
{
  int n = blockIdx.y, c = blockIdx.x;
  const float* p = x + ((long)n * C + c) * T * V;
  float s = 0.f;
  for (int i = threadIdx.x; i < T * V; i += 256) s += p[i];
  __shared__ float r[256];
  r[threadIdx.x] = s; __syncthreads();
  for (int st = 128; st > 0; st >>= 1) {
    if (threadIdx.x < st) r[threadIdx.x] += r[threadIdx.x + st];
    __syncthreads();
  }
  if (threadIdx.x == 0) feat[n * C + c] = r[0] / (float)(T * V);
}

__global__ void k_fc_softmax(const float* __restrict__ feat, const float* __restrict__ fw,
                             const float* __restrict__ fb, float* __restrict__ out)
{
  int n = blockIdx.x, tid = threadIdx.x;
  __shared__ float l[256];
  __shared__ float mx, sm;
  if (tid < 249) {
    float s = fb[tid];
    for (int c = 0; c < 256; ++c) s += fw[tid * 256 + c] * feat[n * 256 + c];
    l[tid] = s;
  }
  __syncthreads();
  if (tid == 0) {
    float m = -1e30f;
    for (int i = 0; i < 249; ++i) m = fmaxf(m, l[i]);
    float s = 0.f;
    for (int i = 0; i < 249; ++i) s += expf(l[i] - m);
    mx = m; sm = s;
  }
  __syncthreads();
  if (tid < 249) out[n * 249 + tid] = expf(l[tid] - mx) / sm;
}

// ---------------- Host orchestration -------------------------------
extern "C" void kernel_launch(void* const* d_in, const int* in_sizes, int n_in,
                              void* d_out, int out_size, void* d_ws, size_t ws_size,
                              hipStream_t stream)
{
  (void)in_sizes; (void)n_in; (void)out_size; (void)ws_size;
  int ip = 0;
  auto nxt = [&]() { return (const float*)d_in[ip++]; };

  const float* X = nxt();
  const float* stem_w = nxt(); const float* stem_b = nxt();
  const float* stem_bn[4]; for (int i = 0; i < 4; ++i) stem_bn[i] = nxt();
  struct OffP { const float *thw,*thb,*phw,*phb,*gw,*gb,*ow,*ob,*fw,*fb; } op[4];
  for (int i = 0; i < 4; ++i) {
    op[i].thw = nxt(); op[i].thb = nxt(); op[i].phw = nxt(); op[i].phb = nxt();
    op[i].gw = nxt();  op[i].gb = nxt();  op[i].ow = nxt();  op[i].ob = nxt();
    op[i].fw = nxt();  op[i].fb = nxt();
  }
  const float* sig_bw = nxt(); const float* sig_bb = nxt();
  const float* sig_ow = nxt(); const float* sig_ob = nxt();
  const float* sig_bn[4]; for (int i = 0; i < 4; ++i) sig_bn[i] = nxt();
  const float* sf_w = nxt(); const float* sf_b = nxt();
  const float* sf_bn[4]; for (int i = 0; i < 4; ++i) sf_bn[i] = nxt();
  const float* nf_w = nxt(); const float* nf_b = nxt();
  const float* nf_bn[4]; for (int i = 0; i < 4; ++i) nf_bn[i] = nxt();
  const float* dbn[4];   for (int i = 0; i < 4; ++i) dbn[i] = nxt();

  static const int CI[10] = {64,64,64,64,64,128,128,128,256,256};
  static const int CO[10] = {64,64,64,64,128,128,128,256,256,256};
  static const int ST[10] = {1,1,1,1,2,1,1,2,1,1};
  static const int RM[10] = {0,1,1,1,2,1,1,2,1,1};  // 0=no res, 1=identity, 2=conv
  struct Lay { const float *gw,*gb,*b1g,*b1b,*b1m,*b1v,*tw,*tb,*b2g,*b2b,*b2m,*b2v,*edge,
                           *rw,*rb,*rg,*rbb,*rm,*rv; } L[10];
  for (int i = 0; i < 10; ++i) {
    L[i].gw = nxt(); L[i].gb = nxt();
    L[i].b1g = nxt(); L[i].b1b = nxt(); L[i].b1m = nxt(); L[i].b1v = nxt();
    L[i].tw = nxt(); L[i].tb = nxt();
    L[i].b2g = nxt(); L[i].b2b = nxt(); L[i].b2m = nxt(); L[i].b2v = nxt();
    L[i].edge = nxt();
    if (RM[i] == 2) {
      L[i].rw = nxt(); L[i].rb = nxt();
      L[i].rg = nxt(); L[i].rbb = nxt(); L[i].rm = nxt(); L[i].rv = nxt();
    } else { L[i].rw = L[i].rb = L[i].rg = L[i].rbb = L[i].rm = L[i].rv = nullptr; }
  }
  const float* fcw = nxt(); const float* fcb = nxt();

  // ---- workspace layout (bump + explicit region reuse) ----
  float* Wf = (float*)d_ws;
  size_t oo = 0;
  auto alloc = [&](size_t n) { float* p = Wf + oo; oo += n; return p; };
  const long S = 44L * 512;
  float* h    = alloc(8 * 32 * S);
  float* th   = alloc(8 * 32 * S);     // reused: z3
  float* ph   = alloc(8 * 32 * S);     // reused (with gg/yb/offb region): sig39
  float* gg   = alloc(8 * 32 * S);
  float* yb   = alloc(8 * 32 * S);
  float* offb = alloc(8 * 64 * S);
  float* xos  = alloc(32 * 32 * S);    // reused: h1/yg/res backbone scratch
  float* nlre = alloc(8 * 128 * S);
  float* spre = alloc(8 * 128 * S);    // reused: bxB ping-pong buffer
  float* spf  = alloc(8 * 32 * S);
  float* nlf  = alloc(8 * 32 * S);
  float* bxA  = alloc(8L * 64 * 512 * 44);
  float* Ae   = alloc(3 * 44 * 44);
  float* feat = alloc(8 * 256);
  float* z3  = th;
  float* s39 = ph;                      // needs 32*39*S <= (ph..offb) = 1280*S  OK
  float* h1  = xos;                     // per-n: max 384*512*44
  float* yg  = xos + 384L * 512 * 44;   // per-n: max 128*512*44
  float* res = yg  + 128L * 512 * 44;   // per-n: max 128*256*44
  float* bxB = spre;

  auto conv = [&](const float* Wm, const float* Xin, float* Outp,
                  const float* bias, const float* g4, const float* b4,
                  const float* m4, const float* v4, const float* Rp,
                  int M, int Cin, int ntaps, int pad, int stride,
                  int Tin, int Vd, int Tout, int flags) {
    int Ncols = Tout * Vd;
    dim3 grid((Ncols + 127) / 128, (M + 63) / 64, 1);
    k_conv_wmma<<<grid, 256, 0, stream>>>(Wm, Xin, Outp, bias, g4, b4, m4, v4, Rp,
                                          M, Cin, ntaps, pad, stride, Tin, Vd, Ncols, flags);
  };

  // ---- stem ----
  {
    int tot = (int)(8 * 32 * S);
    k_stem<<<(tot + 255) / 256, 256, 0, stream>>>(
        X, stem_w, stem_b, stem_bn[0], stem_bn[1], stem_bn[2], stem_bn[3], h, tot);
  }

  // ---- 4 offset blocks ----
  for (int i = 0; i < 4; ++i) {
    for (int b = 0; b < 8; ++b) {
      conv(op[i].thw, h + b * 32 * S, th + b * 32 * S, op[i].thb, 0,0,0,0, nullptr,
           32, 32, 1, 0, 1, (int)S, 1, (int)S, EF_BIAS);
      conv(op[i].phw, h + b * 32 * S, ph + b * 32 * S, op[i].phb, 0,0,0,0, nullptr,
           32, 32, 1, 0, 1, (int)S, 1, (int)S, EF_BIAS);
      conv(op[i].gw,  h + b * 32 * S, gg + b * 32 * S, op[i].gb, 0,0,0,0, nullptr,
           32, 32, 1, 0, 1, (int)S, 1, (int)S, EF_BIAS);
    }
    k_attn<<<4096, 256, 0, stream>>>(th, ph, gg, yb);
    for (int b = 0; b < 8; ++b) {
      // x_nl written directly into regrouped (b, 128) layout: channel block i*32
      conv(op[i].ow, yb + b * 32 * S, nlre + ((long)b * 128 + i * 32) * S, op[i].ob,
           0,0,0,0, nullptr, 32, 32, 1, 0, 1, (int)S, 1, (int)S, EF_BIAS);
      conv(op[i].fw, h + b * 32 * S, offb + b * 64 * S, op[i].fb,
           0,0,0,0, nullptr, 64, 32, 1, 0, 1, (int)S, 1, (int)S, EF_BIAS);
    }
    int tot = (int)(8 * 32 * S);
    k_deform<<<(tot + 255) / 256, 256, 0, stream>>>(h, offb, xos + (long)i * tot, tot);
  }

  // ---- signature path ----
  {
    int tot = (int)(32 * 3 * S);
    k_bott3<<<(tot + 255) / 256, 256, 0, stream>>>(xos, sig_bw, sig_bb, z3, tot);
    tot = (int)(32 * S);
    k_sig<<<(tot + 255) / 256, 256, 0, stream>>>(z3, s39, tot);
  }
  for (int n = 0; n < 32; ++n) {  // sig out conv, written into regrouped (b,128) layout
    conv(sig_ow, s39 + (long)n * 39 * S,
         spre + (((long)(n & 7)) * 128 + (n >> 3) * 32) * S, sig_ob,
         sig_bn[0], sig_bn[1], sig_bn[2], sig_bn[3], nullptr,
         32, 39, 1, 0, 1, (int)S, 1, (int)S, EF_BIAS | EF_BN | EF_RELU);
  }
  for (int b = 0; b < 8; ++b) {  // fusions
    conv(sf_w, spre + (long)b * 128 * S, spf + (long)b * 32 * S, sf_b,
         sf_bn[0], sf_bn[1], sf_bn[2], sf_bn[3], nullptr,
         32, 128, 1, 0, 1, (int)S, 1, (int)S, EF_BIAS | EF_BN | EF_RELU);
    conv(nf_w, nlre + (long)b * 128 * S, nlf + (long)b * 32 * S, nf_b,
         nf_bn[0], nf_bn[1], nf_bn[2], nf_bn[3], nullptr,
         32, 128, 1, 0, 1, (int)S, 1, (int)S, EF_BIAS | EF_BN | EF_RELU);
  }

  // ---- build backbone input (transpose + concat + data_bn) ----
  {
    int tot = (int)(8L * 64 * 512 * 44);
    k_build_bx<<<(tot + 255) / 256, 256, 0, stream>>>(
        h, nlf, spf, dbn[0], dbn[1], dbn[2], dbn[3], bxA, tot);
  }

  // ---- backbone: 10 ST-GCN blocks, per-batch loop ----
  float* xc = bxA; float* xn = bxB;
  int Tc = 512;
  for (int li = 0; li < 10; ++li) {
    int ci = CI[li], co = CO[li], st = ST[li];
    int To = (Tc - 1) / st + 1;
    k_make_adj<<<(5808 + 255) / 256, 256, 0, stream>>>(L[li].edge, Ae);
    for (int n = 0; n < 8; ++n) {
      const float* xin = xc + (long)n * ci * Tc * 44;
      conv(L[li].gw, xin, h1, L[li].gb, 0,0,0,0, nullptr,
           3 * co, ci, 1, 0, 1, Tc, 44, Tc, EF_BIAS);
      {
        dim3 g(1, (co * Tc + 63) / 64, 1);
        k_graph_wmma<<<g, 256, 0, stream>>>(h1, Ae, yg,
            L[li].b1g, L[li].b1b, L[li].b1m, L[li].b1v,
            co * Tc, Tc, co * Tc * 44);
      }
      const float* resp = nullptr;
      int fl = EF_BIAS | EF_BN | EF_RELU;
      if (RM[li] == 2) {
        conv(L[li].rw, xin, res, L[li].rb,
             L[li].rg, L[li].rbb, L[li].rm, L[li].rv, nullptr,
             co, ci, 1, 0, st, Tc, 44, To, EF_BIAS | EF_BN);
        resp = res; fl |= EF_RES;
      } else if (RM[li] == 1) {
        resp = xin; fl |= EF_RES;  // identity: ci==co, st==1
      }
      conv(L[li].tw, yg, xn + (long)n * co * To * 44, L[li].tb,
           L[li].b2g, L[li].b2b, L[li].b2m, L[li].b2v, resp,
           co, co, 9, 4, st, Tc, 44, To, fl);
    }
    float* tmp = xc; xc = xn; xn = tmp;
    Tc = To;
  }

  // ---- global mean pool + fc + softmax ----
  k_mean<<<dim3(256, 8), 256, 0, stream>>>(xc, feat, 256, Tc, 44);
  k_fc_softmax<<<8, 256, 0, stream>>>(feat, fcw, fcb, (float*)d_out);
}
